// GCN_33423435497638
// MI455X (gfx1250) — matrix operand order repaired and checked by reading, not compiled, not run on hardware
//
#include <hip/hip_runtime.h>
#include <hip/hip_bf16.h>
#include <math.h>

// ---------------------------------------------------------------------------
// GAT 2-layer forward for MI455X (gfx1250, wave32, WMMA).
// GEMMs run on v_wmma_f32_16x16x32_f16 with fragment-packed operands.
// ---------------------------------------------------------------------------

typedef __attribute__((ext_vector_type(16))) _Float16 v16h;
typedef __attribute__((ext_vector_type(8)))  float    v8f;

#define IN_CH   128
#define HID     32
#define HEADS   8
#define OUT_CH  40
#define C1      (HEADS * HID)   // 256
#define NPAD2   48              // OUT_CH padded to n-tile multiple
#define NEG_SLOPE 0.2f

// ---------------------------- utility kernels ------------------------------

__global__ __launch_bounds__(256) void fill_f32(float* p, float v, int n) {
    int i = blockIdx.x * blockDim.x + threadIdx.x;
    if (i < n) p[i] = v;
}

// Pack A matrix (f32 [M,K] -> f16 fragment layout).
// Layout: [m_tile][k_chunk][lane(32)][16 halves]
//   lane<16 : row = mt*16+lane,    kb = kc*32 + 0
//   lane>=16: row = mt*16+lane-16, kb = kc*32 + 8
//   element e<8  -> K = kb + e        (VGPR 0..3)
//   element e>=8 -> K = kb + 8 + e    (VGPR 4..7, i.e. kb+16..kb+23)
__global__ __launch_bounds__(256) void pack_a(const float* __restrict__ src,
                                              _Float16* __restrict__ dst,
                                              int M, int K, int m_tiles, int kch) {
    int id = blockIdx.x * blockDim.x + threadIdx.x;
    int total = m_tiles * kch * 32 * 16;
    if (id >= total) return;
    int e    = id & 15;
    int lane = (id >> 4) & 31;
    int kc   = (id >> 9) % kch;
    int mt   = (id >> 9) / kch;
    int row  = mt * 16 + (lane & 15);
    int kb   = kc * 32 + ((lane >> 4) * 8);
    int k    = (e < 8) ? (kb + e) : (kb + 8 + e);
    float v  = (row < M) ? src[(size_t)row * K + k] : 0.0f;
    dst[id] = (_Float16)v;
}

// Pack B matrix (weights f32 [K,Nsrc] -> f16 fragment layout).
// Layout: [n_tile][k_chunk][lane(32)][16 halves]
//   col = nt*16 + (lane&15);  K = kc*32 + (lane>>4)*8 + (j<8 ? j : j+8)  (j = element idx)
__global__ __launch_bounds__(256) void pack_b(const float* __restrict__ src,
                                              _Float16* __restrict__ dst,
                                              int K, int Nsrc, int n_tiles, int kch) {
    int id = blockIdx.x * blockDim.x + threadIdx.x;
    int total = n_tiles * kch * 32 * 16;
    if (id >= total) return;
    int j    = id & 15;
    int lane = (id >> 4) & 31;
    int kc   = (id >> 9) % kch;
    int nt   = (id >> 9) / kch;
    int col  = nt * 16 + (lane & 15);
    int k    = kc * 32 + ((lane >> 4) * 8) + ((j < 8) ? j : (j + 8));
    float v  = (col < Nsrc && k < K) ? src[(size_t)k * Nsrc + col] : 0.0f;
    dst[id] = (_Float16)v;
}

// ------------------------------ WMMA GEMM ----------------------------------
// One wave per 16x16 output tile; 8 waves per 256-thread block.
// A/B operands pre-packed so each lane issues contiguous 32B fragment loads.
__global__ __launch_bounds__(256) void gemm_wmma(const _Float16* __restrict__ A,
                                                 const _Float16* __restrict__ B,
                                                 float* __restrict__ C,
                                                 int m_tiles, int n_tiles, int kch,
                                                 int ldc) {
    int wave = threadIdx.x >> 5;
    int lane = threadIdx.x & 31;
    int tile = blockIdx.x * 8 + wave;
    if (tile >= m_tiles * n_tiles) return;   // uniform per wave
    int mt = tile / n_tiles;
    int nt = tile % n_tiles;

    v8f acc;
#pragma unroll
    for (int i = 0; i < 8; ++i) acc[i] = 0.0f;

    const size_t abase = ((size_t)mt * kch) * 32 * 16 + (size_t)lane * 16;
    const size_t bbase = ((size_t)nt * kch) * 32 * 16 + (size_t)lane * 16;
    for (int kc = 0; kc < kch; ++kc) {
        v16h a = *reinterpret_cast<const v16h*>(A + abase + (size_t)kc * 32 * 16);
        v16h b = *reinterpret_cast<const v16h*>(B + bbase + (size_t)kc * 32 * 16);
        acc = __builtin_amdgcn_wmma_f32_16x16x32_f16(false, a, false, b,
                                                     (short)0, acc, false, false);
    }

    // D layout: VGPR r -> row r (lanes 0-15) / row r+8 (lanes 16-31), col = lane&15
    int col = nt * 16 + (lane & 15);
    int rbase = mt * 16 + ((lane >> 4) * 8);
#pragma unroll
    for (int r = 0; r < 8; ++r)
        C[(size_t)(rbase + r) * ldc + col] = acc[r];
}

// --------------------------- attention pieces ------------------------------

__global__ __launch_bounds__(256) void att_coeff(const float* __restrict__ xp,
                                                 const float* __restrict__ att_s,
                                                 const float* __restrict__ att_d,
                                                 float* __restrict__ a_src,
                                                 float* __restrict__ a_dst,
                                                 int N, int H, int Cc, int ld) {
    int id = blockIdx.x * blockDim.x + threadIdx.x;
    if (id >= N * H) return;
    int h = id % H, n = id / H;
    const float* row = xp + (size_t)n * ld + h * Cc;
    const float* as = att_s + h * Cc;
    const float* ad = att_d + h * Cc;
    float s = 0.f, d = 0.f;
    for (int c = 0; c < Cc; ++c) { float v = row[c]; s += v * as[c]; d += v * ad[c]; }
    a_src[id] = s;
    a_dst[id] = d;
}

__device__ __forceinline__ void atomicMaxF(float* addr, float v) {
    if (v >= 0.0f) atomicMax((int*)addr, __float_as_int(v));
    else           atomicMin((unsigned int*)addr, (unsigned int)__float_as_int(v));
}

__device__ __forceinline__ void edge_nodes(const int* ei, int e, int E, int& s, int& d) {
    if (e < E) { s = ei[e]; d = ei[E + e]; }
    else       { s = d = e - E; }   // self loop
}

__global__ __launch_bounds__(256) void edge_max(const float* __restrict__ a_src,
                                                const float* __restrict__ a_dst,
                                                const int* __restrict__ ei,
                                                int E, int N, int H,
                                                float* __restrict__ m) {
    int id = blockIdx.x * blockDim.x + threadIdx.x;
    int total = (E + N) * H;
    if (id >= total) return;
    int h = id % H, e = id / H;
    int s, d; edge_nodes(ei, e, E, s, d);
    float l = a_src[s * H + h] + a_dst[d * H + h];
    l = (l > 0.f) ? l : NEG_SLOPE * l;
    atomicMaxF(&m[d * H + h], l);
}

__global__ __launch_bounds__(256) void edge_exp(const float* __restrict__ a_src,
                                                const float* __restrict__ a_dst,
                                                const int* __restrict__ ei,
                                                int E, int N, int H,
                                                const float* __restrict__ m,
                                                float* __restrict__ p,
                                                float* __restrict__ z) {
    int id = blockIdx.x * blockDim.x + threadIdx.x;
    int total = (E + N) * H;
    if (id >= total) return;
    int h = id % H, e = id / H;
    int s, d; edge_nodes(ei, e, E, s, d);
    float l = a_src[s * H + h] + a_dst[d * H + h];
    l = (l > 0.f) ? l : NEG_SLOPE * l;
    float pe = __expf(l - m[d * H + h]);
    p[id] = pe;
    atomicAdd(&z[d * H + h], pe);
}

__global__ __launch_bounds__(256) void edge_agg(const float* __restrict__ p,
                                                const float* __restrict__ z,
                                                const float* __restrict__ xp,
                                                const int* __restrict__ ei,
                                                int E, int N, int H, int Cc, int ldx,
                                                float* __restrict__ out) {
    int cht = H * Cc;
    int id = blockIdx.x * blockDim.x + threadIdx.x;
    long total = (long)(E + N) * cht;
    if ((long)id >= total) return;
    int c = id % cht, e = id / cht;
    int h = c / Cc;
    int s, d; edge_nodes(ei, e, E, s, d);
    float alpha = p[(size_t)e * H + h] / (z[d * H + h] + 1e-16f);
    atomicAdd(&out[(size_t)d * cht + c], alpha * xp[(size_t)s * ldx + c]);
}

// ------------------------------ epilogues ----------------------------------

__global__ __launch_bounds__(256) void relu_bias(float* __restrict__ h,
                                                 const float* __restrict__ b,
                                                 int N, int Cc) {
    int id = blockIdx.x * blockDim.x + threadIdx.x;
    if (id >= N * Cc) return;
    int c = id % Cc;
    h[id] = fmaxf(h[id] + b[c], 0.0f);
}

__global__ __launch_bounds__(128) void bias_logsoftmax(const float* __restrict__ agg,
                                                       const float* __restrict__ b,
                                                       float* __restrict__ out,
                                                       int N) {
    int n = blockIdx.x * blockDim.x + threadIdx.x;
    if (n >= N) return;
    float v[OUT_CH];
    float mx = -INFINITY;
    for (int c = 0; c < OUT_CH; ++c) {
        v[c] = agg[(size_t)n * OUT_CH + c] + b[c];
        mx = fmaxf(mx, v[c]);
    }
    float sum = 0.f;
    for (int c = 0; c < OUT_CH; ++c) sum += __expf(v[c] - mx);
    float lse = __logf(sum);
    for (int c = 0; c < OUT_CH; ++c)
        out[(size_t)n * OUT_CH + c] = v[c] - mx - lse;
}

// ------------------------------- launcher ----------------------------------

static inline int grid1(long total, int block) { return (int)((total + block - 1) / block); }

extern "C" void kernel_launch(void* const* d_in, const int* in_sizes, int n_in,
                              void* d_out, int out_size, void* d_ws, size_t ws_size,
                              hipStream_t stream) {
    const float* x        = (const float*)d_in[0];
    const int*   ei       = (const int*)  d_in[1];
    const float* W1       = (const float*)d_in[2];
    const float* att_s1   = (const float*)d_in[3];
    const float* att_d1   = (const float*)d_in[4];
    const float* b1       = (const float*)d_in[5];
    const float* W2       = (const float*)d_in[6];
    const float* att_s2   = (const float*)d_in[7];
    const float* att_d2   = (const float*)d_in[8];
    const float* b2       = (const float*)d_in[9];

    const int N = in_sizes[0] / IN_CH;     // 100000
    const int E = in_sizes[1] / 2;         // 800000
    const int Etot = E + N;                // with self loops
    const int m_tiles = (N + 15) / 16;
    const int Mpad = m_tiles * 16;

    // GEMM1: K=128 (4 k-chunks), Npad=256 (16 n-tiles)
    // GEMM2: K=256 (8 k-chunks), Npad=48  (3 n-tiles)
    const int KCH1 = IN_CH / 32, NT1 = C1 / 16;
    const int KCH2 = C1 / 32,    NT2 = NPAD2 / 16;

    // ---- workspace layout ----
    size_t cur = 0;
    auto alloc = [&](size_t bytes) {
        size_t off = cur; cur += (bytes + 255) & ~(size_t)255; return off;
    };
    char* ws = (char*)d_ws;
    _Float16* Apack1 = (_Float16*)(ws + alloc((size_t)Mpad * IN_CH * 2));
    _Float16* Bpack1 = (_Float16*)(ws + alloc((size_t)NT1 * KCH1 * 32 * 16 * 2));
    float*    xp1    = (float*)   (ws + alloc((size_t)Mpad * C1 * 4));
    float*    asrc1  = (float*)   (ws + alloc((size_t)N * HEADS * 4));
    float*    adst1  = (float*)   (ws + alloc((size_t)N * HEADS * 4));
    float*    m1     = (float*)   (ws + alloc((size_t)N * HEADS * 4));
    float*    z1     = (float*)   (ws + alloc((size_t)N * HEADS * 4));
    float*    p1     = (float*)   (ws + alloc((size_t)Etot * HEADS * 4));
    float*    hbuf   = (float*)   (ws + alloc((size_t)Mpad * C1 * 4));
    _Float16* Apack2 = (_Float16*)(ws + alloc((size_t)Mpad * C1 * 2));
    _Float16* Bpack2 = (_Float16*)(ws + alloc((size_t)NT2 * KCH2 * 32 * 16 * 2));
    float*    xp2    = (float*)   (ws + alloc((size_t)Mpad * NPAD2 * 4));
    float*    asrc2  = (float*)   (ws + alloc((size_t)N * 4));
    float*    adst2  = (float*)   (ws + alloc((size_t)N * 4));
    float*    m2     = (float*)   (ws + alloc((size_t)N * 4));
    float*    z2     = (float*)   (ws + alloc((size_t)N * 4));
    float*    p2     = (float*)   (ws + alloc((size_t)Etot * 4));
    float*    agg2   = (float*)   (ws + alloc((size_t)N * OUT_CH * 4));
    (void)ws_size;

    const int B = 256;

    // ---- init accumulators ----
    fill_f32<<<grid1((long)N * HEADS, B), B, 0, stream>>>(m1, -INFINITY, N * HEADS);
    fill_f32<<<grid1((long)N * HEADS, B), B, 0, stream>>>(z1, 0.0f, N * HEADS);
    fill_f32<<<grid1((long)Mpad * C1, B), B, 0, stream>>>(hbuf, 0.0f, Mpad * C1);
    fill_f32<<<grid1(N, B), B, 0, stream>>>(m2, -INFINITY, N);
    fill_f32<<<grid1(N, B), B, 0, stream>>>(z2, 0.0f, N);
    fill_f32<<<grid1((long)N * OUT_CH, B), B, 0, stream>>>(agg2, 0.0f, N * OUT_CH);

    // ---- pack weights ----
    pack_b<<<grid1((long)NT1 * KCH1 * 512, B), B, 0, stream>>>(W1, Bpack1, IN_CH, C1, NT1, KCH1);
    pack_b<<<grid1((long)NT2 * KCH2 * 512, B), B, 0, stream>>>(W2, Bpack2, C1, OUT_CH, NT2, KCH2);

    // ======== layer 1 ========
    pack_a<<<grid1((long)m_tiles * KCH1 * 512, B), B, 0, stream>>>(x, Apack1, N, IN_CH, m_tiles, KCH1);
    gemm_wmma<<<grid1((long)m_tiles * NT1, 8), B, 0, stream>>>(Apack1, Bpack1, xp1,
                                                               m_tiles, NT1, KCH1, C1);
    att_coeff<<<grid1((long)N * HEADS, B), B, 0, stream>>>(xp1, att_s1, att_d1,
                                                           asrc1, adst1, N, HEADS, HID, C1);
    edge_max<<<grid1((long)Etot * HEADS, B), B, 0, stream>>>(asrc1, adst1, ei, E, N, HEADS, m1);
    edge_exp<<<grid1((long)Etot * HEADS, B), B, 0, stream>>>(asrc1, adst1, ei, E, N, HEADS,
                                                             m1, p1, z1);
    edge_agg<<<grid1((long)Etot * C1, B), B, 0, stream>>>(p1, z1, xp1, ei, E, N,
                                                          HEADS, HID, C1, hbuf);
    relu_bias<<<grid1((long)N * C1, B), B, 0, stream>>>(hbuf, b1, N, C1);

    // ======== layer 2 ========
    pack_a<<<grid1((long)m_tiles * KCH2 * 512, B), B, 0, stream>>>(hbuf, Apack2, N, C1, m_tiles, KCH2);
    gemm_wmma<<<grid1((long)m_tiles * NT2, 8), B, 0, stream>>>(Apack2, Bpack2, xp2,
                                                               m_tiles, NT2, KCH2, NPAD2);
    att_coeff<<<grid1(N, B), B, 0, stream>>>(xp2, att_s2, att_d2, asrc2, adst2,
                                             N, 1, OUT_CH, NPAD2);
    edge_max<<<grid1(Etot, B), B, 0, stream>>>(asrc2, adst2, ei, E, N, 1, m2);
    edge_exp<<<grid1(Etot, B), B, 0, stream>>>(asrc2, adst2, ei, E, N, 1, m2, p2, z2);
    edge_agg<<<grid1((long)Etot * OUT_CH, B), B, 0, stream>>>(p2, z2, xp2, ei, E, N,
                                                              1, OUT_CH, NPAD2, agg2);
    bias_logsoftmax<<<grid1(N, 128), 128, 0, stream>>>(agg2, b2, (float*)d_out, N);
}
